// _LSTM_35871566856225
// MI455X (gfx1250) — compile-verified, matrix operand order checked by reading, not run on hardware
//
#include <hip/hip_runtime.h>

// T=1 LSTM: every layer is a single cell step with h0=c0=0, so
//   - all w_hh terms vanish (h0 = 0)
//   - the forget gate is dead (c0 = 0)
// Per layer: gates = in @ w_ih.T + b; c = sig(i)*tanh(g); h = sig(o)*tanh(c)
// Final: out = h @ fc_w.T + fc_b.
// GEMMs: v_wmma_f32_16x16x32_f16, 16 batch rows per wave.
// Activations: gfx1250 native v_tanh_f32; sig(x) = 0.5 + 0.5*tanh(x/2),
// with biases pre-scaled so they fold into the sigmoid input fma.

#define NW    4                 // waves per block (wave32)
#define TPB   (NW * 32)
#define NBLK  1024

typedef _Float16 v8h  __attribute__((ext_vector_type(8)));
typedef _Float16 v16h __attribute__((ext_vector_type(16)));
typedef float    v8f  __attribute__((ext_vector_type(8)));
typedef float    v4f  __attribute__((ext_vector_type(4)));

static constexpr int BATCH = 524288;
static constexpr int E  = 64;   // embed dim
static constexpr int H  = 32;   // hidden
static constexpr int GC = 96;   // compact gate rows kept: i(0..31), g(32..63), o(64..95)

__device__ __forceinline__ v8f wmma_f16(v16h a, v16h b, v8f c) {
  return __builtin_amdgcn_wmma_f32_16x16x32_f16(false, a, false, b, (short)0, c,
                                                false, false);
}

// sigmoid(x + b) with hb = 0.5*b prestaged: 0.5 + 0.5*tanh(0.5*x + hb)
__device__ __forceinline__ float sig_b(float x, float hb) {
  return __builtin_fmaf(__builtin_amdgcn_tanhf(__builtin_fmaf(x, 0.5f, hb)),
                        0.5f, 0.5f);
}

// 16-bit fragment chunking (per ISA A-layout, mirrored for B):
// lane: m16 = lane&15 selects row (A) / column (B); g = lane>>4.
// halves 0..7  -> K = 8*g + h ; halves 8..15 -> K = 16 + 8*g + (h-8)
// => two contiguous 8-half (16B) chunks at offsets 8g and 16+8g.
__device__ __forceinline__ v16h frag_lds(const _Float16* W, int ldk, int n0,
                                         int k0, int m16, int g) {
  const _Float16* p = W + (n0 + m16) * ldk + k0 + 8 * g;
  v8h lo = *(const v8h*)(p);
  v8h hi = *(const v8h*)(p + 16);
  return __builtin_shufflevector(lo, hi, 0, 1, 2, 3, 4, 5, 6, 7,
                                 8, 9, 10, 11, 12, 13, 14, 15);
}

// A fragment straight from global f32 x (row stride 64 floats), converted to f16.
__device__ __forceinline__ v16h fragA_global(const float* xb, int k0, int m16,
                                             int g) {
  const float* p = xb + m16 * 64 + k0 + 8 * g;
  v4f f0 = *(const v4f*)(p);
  v4f f1 = *(const v4f*)(p + 4);
  v4f f2 = *(const v4f*)(p + 16);
  v4f f3 = *(const v4f*)(p + 20);
  v16h r;
#pragma unroll
  for (int i = 0; i < 4; ++i) {
    r[i]      = (_Float16)f0[i];
    r[4 + i]  = (_Float16)f1[i];
    r[8 + i]  = (_Float16)f2[i];
    r[12 + i] = (_Float16)f3[i];
  }
  return r;
}

__global__ __launch_bounds__(TPB) void lstm_fused_wmma(
    const float* __restrict__ x,          // (B, 64)
    const float* __restrict__ w_ih0,      // (128, 64)
    const float* __restrict__ w_ih_rest,  // (4, 128, 32)
    const float* __restrict__ b_ih,       // (5, 128)
    const float* __restrict__ b_hh,       // (5, 128)
    const float* __restrict__ fc_w,       // (64, 32)
    const float* __restrict__ fc_b,       // (64,)
    float* __restrict__ out)              // (B, 64)
{
  __shared__ __align__(16) _Float16 sW0[GC * 64];      // layer 0 weights (i,g,o rows)
  __shared__ __align__(16) _Float16 sWr[4][GC * 32];   // layers 1..4
  __shared__ __align__(16) _Float16 sFC[E * H];        // fc weights (64x32)
  __shared__ float sB[5][GC];   // biases: i rows scaled 0.5, g rows raw, o rows scaled 0.5
  __shared__ float sFB[E];
  __shared__ __align__(16) _Float16 sH[NW][16 * 32];   // per-wave hidden swizzle buf

  const int tid = threadIdx.x;

  // ---- stage weights (f32 -> f16, compact i/g/o rows) ----
  for (int i = tid; i < GC * 64; i += TPB) {
    int r = i >> 6, k = i & 63;
    int ro = (r < 32) ? r : r + 32;                    // skip forget rows 32..63
    sW0[i] = (_Float16)w_ih0[ro * 64 + k];
  }
#pragma unroll
  for (int l = 0; l < 4; ++l)
    for (int i = tid; i < GC * 32; i += TPB) {
      int r = i >> 5, k = i & 31;
      int ro = (r < 32) ? r : r + 32;
      sWr[l][i] = (_Float16)w_ih_rest[(l * 128 + ro) * 32 + k];
    }
  for (int i = tid; i < E * H; i += TPB) sFC[i] = (_Float16)fc_w[i];
  for (int i = tid; i < 5 * GC; i += TPB) {
    int l = i / GC, r = i % GC;
    int ro = (r < 32) ? r : r + 32;
    float bv = b_ih[l * 128 + ro] + b_hh[l * 128 + ro];
    // i and o gate biases pre-scaled by 0.5 for the sigmoid input fma;
    // g gate bias kept raw for tanh(g + b).
    sB[l][r] = (r < 32 || r >= 64) ? 0.5f * bv : bv;
  }
  for (int i = tid; i < E; i += TPB) sFB[i] = fc_b[i];
  __syncthreads();

  const int wid = tid >> 5, lane = tid & 31;
  const int m16 = lane & 15, g = lane >> 4;
  _Float16* hb = sH[wid];

  v8f zero;
#pragma unroll
  for (int i = 0; i < 8; ++i) zero[i] = 0.0f;   // loop-invariant C init

  const int nTiles = BATCH / 16;
  const int gw = blockIdx.x * NW + wid;
  const int nWaves = gridDim.x * NW;

  for (int tile = gw; tile < nTiles; tile += nWaves) {
    const int row0 = tile * 16;
    const float* xb = x + (size_t)row0 * E;

    // ---------- layer 0: A = x tile (16x64), two K-steps ----------
    v16h a0 = fragA_global(xb, 0, m16, g);
    v16h a1 = fragA_global(xb, 32, m16, g);
    v8f acc[6];
#pragma unroll
    for (int t = 0; t < 6; ++t) {
      v8f c = wmma_f16(a0, frag_lds(sW0, 64, t * 16, 0, m16, g), zero);
      acc[t] = wmma_f16(a1, frag_lds(sW0, 64, t * 16, 32, m16, g), c);
    }
    // activations: acc[0..1]=i, acc[2..3]=gbar, acc[4..5]=o
#pragma unroll
    for (int half = 0; half < 2; ++half) {
      float bi = sB[0][half * 16 + m16];
      float bg = sB[0][32 + half * 16 + m16];
      float bo = sB[0][64 + half * 16 + m16];
      v8f gi = acc[half], gg = acc[2 + half], go = acc[4 + half];
#pragma unroll
      for (int r = 0; r < 8; ++r) {
        float cv = sig_b(gi[r], bi) * __builtin_amdgcn_tanhf(gg[r] + bg);
        float hv = sig_b(go[r], bo) * __builtin_amdgcn_tanhf(cv);
        int m = g * 8 + r;                       // C layout: M = vgpr + 8*(lane/16)
        hb[m * 32 + half * 16 + m16] = (_Float16)hv;
      }
    }

    // ---------- layers 1..4: A = hidden (16x32), one K-step ----------
#pragma unroll
    for (int l = 0; l < 4; ++l) {
      v16h a = frag_lds(hb, 32, 0, 0, m16, g);
#pragma unroll
      for (int t = 0; t < 6; ++t)
        acc[t] = wmma_f16(a, frag_lds(sWr[l], 32, t * 16, 0, m16, g), zero);
#pragma unroll
      for (int half = 0; half < 2; ++half) {
        float bi = sB[l + 1][half * 16 + m16];
        float bg = sB[l + 1][32 + half * 16 + m16];
        float bo = sB[l + 1][64 + half * 16 + m16];
        v8f gi = acc[half], gg = acc[2 + half], go = acc[4 + half];
#pragma unroll
        for (int r = 0; r < 8; ++r) {
          float cv = sig_b(gi[r], bi) * __builtin_amdgcn_tanhf(gg[r] + bg);
          float hv = sig_b(go[r], bo) * __builtin_amdgcn_tanhf(cv);
          int m = g * 8 + r;
          hb[m * 32 + half * 16 + m16] = (_Float16)hv;
        }
      }
    }

    // ---------- fc: (16x32) @ (32x64) + b ----------
    v16h a = frag_lds(hb, 32, 0, 0, m16, g);
#pragma unroll
    for (int t = 0; t < 4; ++t) {
      float bv = sFB[t * 16 + m16];
      v8f c = wmma_f16(a, frag_lds(sFC, 32, t * 16, 0, m16, g), zero);
#pragma unroll
      for (int r = 0; r < 8; ++r) {
        int m = g * 8 + r;
        out[(size_t)(row0 + m) * E + t * 16 + m16] = c[r] + bv;
      }
    }
  }
}

extern "C" void kernel_launch(void* const* d_in, const int* in_sizes, int n_in,
                              void* d_out, int out_size, void* d_ws,
                              size_t ws_size, hipStream_t stream) {
  (void)in_sizes; (void)n_in; (void)out_size; (void)d_ws; (void)ws_size;
  const float* x         = (const float*)d_in[0];
  const float* w_ih0     = (const float*)d_in[1];
  // d_in[2] = w_hh0   : dead (h0 == 0, T == 1)
  const float* w_ih_rest = (const float*)d_in[3];
  // d_in[4] = w_hh_rest: dead
  const float* b_ih      = (const float*)d_in[5];
  const float* b_hh      = (const float*)d_in[6];
  const float* fc_w      = (const float*)d_in[7];
  const float* fc_b      = (const float*)d_in[8];
  float* out             = (float*)d_out;

  lstm_fused_wmma<<<dim3(NBLK), dim3(TPB), 0, stream>>>(
      x, w_ih0, w_ih_rest, b_ih, b_hh, fc_w, fc_b, out);
}
